// LAS_18013092839909
// MI455X (gfx1250) — compile-verified, matrix operand order checked by reading, not run on hardware
//
#include <hip/hip_runtime.h>
#include <math.h>

// ---------------------------------------------------------------------------
// Types for CDNA5 WMMA (wave32, v_wmma_f32_16x16x32_f16) and TDM descriptors
// ---------------------------------------------------------------------------
typedef __attribute__((ext_vector_type(16))) _Float16     v16h;
typedef __attribute__((ext_vector_type(8)))  float        v8f;
typedef __attribute__((ext_vector_type(4)))  unsigned int v4u;
typedef __attribute__((ext_vector_type(8)))  int          v8i;
typedef __attribute__((ext_vector_type(4)))  int          v4i;

#define TPB 256
#define KSTEP 64

// Tensor Data Mover path (gfx1250). __has_builtin is false on the host pass,
// so the fallback cooperative-copy staging compiles there automatically.
#if __has_builtin(__builtin_amdgcn_tensor_load_to_lds) && __has_builtin(__builtin_amdgcn_s_wait_tensorcnt)
#define USE_TDM 1
#else
#define USE_TDM 0
#endif

__device__ __forceinline__ float sigf_(float x) { return 1.0f / (1.0f + __expf(-x)); }

#if USE_TDM
// Issue one TDM tensor_load_to_lds: 2D tile (tile_dim0=KSTEP elems, tile_dim1=32
// rows) of 2-byte elements, row stride = K elements, into LDS at ldsOff.
// D# layout per CDNA5 ISA.
__device__ __forceinline__ void tdm_load_a(const _Float16* gsrc, unsigned ldsOff, int K)
{
    unsigned long long ga = (unsigned long long)(size_t)gsrc;
    v4u g0;
    g0.x = 1u;                                              // count=1, user descriptor
    g0.y = ldsOff;                                          // lds_addr
    g0.z = (unsigned)(ga & 0xffffffffu);                    // global_addr[31:0]
    g0.w = (unsigned)((ga >> 32) & 0x1ffffffu) | (2u << 30);// global_addr[56:32], type=2
    v8i g1;
    g1[0] = (1 << 16);                                      // data_size=1 -> 2 bytes
    g1[1] = (int)((K & 0xffff) << 16);                      // tensor_dim0[15:0]
    g1[2] = (int)(((K >> 16) & 0xffff) | (32u << 16));      // tensor_dim0[31:16] | tensor_dim1=32
    g1[3] = (int)(KSTEP << 16);                             // tile_dim0 = KSTEP
    g1[4] = 32;                                             // tile_dim1 = 32, tile_dim2 = 0
    g1[5] = K;                                              // tensor_dim0_stride[31:0]
    g1[6] = 0;                                              // stride[47:32] | dim1_stride lo
    g1[7] = 0;
    v4i z4 = {0, 0, 0, 0};                                  // groups 2/3 unused (2D tensor)
#if __clang_major__ >= 23
    v8i z8 = {0, 0, 0, 0, 0, 0, 0, 0};
    __builtin_amdgcn_tensor_load_to_lds(g0, g1, z4, z4, z8, 0);
#else
    __builtin_amdgcn_tensor_load_to_lds(g0, g1, z4, z4, 0);
#endif
}
#endif

// ---------------------------------------------------------------------------
// Generic WMMA GEMM:  C[M][ldc] = A_f16[M][K] @ W^T  (+ bias)
// Block = 256 thr = 8 waves. Block tile: 32 (M) x 128 (N). Each wave: 1 N-tile,
// 2 M-tiles. A 32xKSTEP slice staged in LDS via TDM (double-buffered, overlapped
// with WMMA; TENSORcnt-tracked) or cooperative copy fallback. B pre-packed so
// each lane loads a contiguous v16h fragment (served from L2 for the serial
// recurrent GEMMs: 1.5 MB weights << 192 MB L2).
// M % 32 == 0, K % KSTEP == 0, Npad % 16 == 0.
// ---------------------------------------------------------------------------
__global__ __launch_bounds__(TPB) void wmma_gemm_kernel(
    const _Float16* __restrict__ A, const _Float16* __restrict__ Bp,
    const float* __restrict__ bias, float* __restrict__ C,
    int M, int N, int Npad, int K, int ldc)
{
    __shared__ __attribute__((aligned(32))) _Float16 Abuf[2][32 * KSTEP];
    const int tid   = threadIdx.x;
    const int wave  = tid >> 5;
    const int lane  = tid & 31;
    const int mBase = blockIdx.y * 32;
    const int ntile = blockIdx.x * 8 + wave;
    const int nBase = ntile * 16;
    const bool active = (nBase < Npad);           // wave-uniform (EXEC all-1 for WMMA)
    const int kT = K / KSTEP;
    const int nlo = lane & 15, khalf = lane >> 4;

    v8f acc0 = {0.f, 0.f, 0.f, 0.f, 0.f, 0.f, 0.f, 0.f};
    v8f acc1 = acc0;

    // B tiles ordered by 32-wide K subtile: ((ntile*(K/32) + kt32)*32 + lane)*16
    const _Float16* Bt = Bp + ((size_t)ntile * (K >> 5)) * 512 + (size_t)lane * 16;
    const _Float16* Arow = A + (size_t)mBase * K;

#if USE_TDM
    if (wave == 0) tdm_load_a(Arow, (unsigned)(size_t)&Abuf[0][0], K);
#endif

    for (int kt = 0; kt < kT; ++kt) {
#if USE_TDM
        if (wave == 0) {
            if (kt + 1 < kT) {
                tdm_load_a(Arow + (kt + 1) * KSTEP,
                           (unsigned)(size_t)&Abuf[(kt + 1) & 1][0], K);
                __builtin_amdgcn_s_wait_tensorcnt(1);   // tile kt landed (in-order)
            } else {
                __builtin_amdgcn_s_wait_tensorcnt(0);
            }
        }
        __syncthreads();                                // publish tile kt
        const _Float16* As = &Abuf[kt & 1][0];
#else
        {   // cooperative stage: 32xKSTEP halves, 16B per thread
            const int idx = tid * 8;
            const int r = idx >> 6, c = idx & 63;
            *(uint4*)&Abuf[0][idx] = *(const uint4*)(Arow + (size_t)r * K + kt * KSTEP + c);
        }
        __syncthreads();
        const _Float16* As = &Abuf[0][0];
#endif
        if (active) {
            #pragma unroll
            for (int ks = 0; ks < 2; ++ks) {
                v16h a0 = *(const v16h*)&As[(nlo)      * KSTEP + ks * 32 + khalf * 16];
                v16h a1 = *(const v16h*)&As[(16 + nlo) * KSTEP + ks * 32 + khalf * 16];
                v16h b  = *(const v16h*)Bt;
                acc0 = __builtin_amdgcn_wmma_f32_16x16x32_f16(
                           false, a0, false, b, (short)0, acc0, false, false);
                acc1 = __builtin_amdgcn_wmma_f32_16x16x32_f16(
                           false, a1, false, b, (short)0, acc1, false, false);
                Bt += 512;
            }
            __builtin_prefetch(Bt, 0, 1);               // global_prefetch_b8 next B tile
        }
        __syncthreads();                                // retire reads of tile kt
    }

    if (active) {
        const int col = nBase + nlo;
        if (col < N) {
            const float bv = bias ? bias[col] : 0.0f;
            const int rowOff = khalf * 8;   // C layout: vgpr r -> M = r + 8*(lane>>4)
            #pragma unroll
            for (int r = 0; r < 8; ++r) {
                const int row = mBase + rowOff + r;
                C[(size_t)row * ldc + col]        = acc0[r] + bv;
                C[(size_t)(row + 16) * ldc + col] = acc1[r] + bv;
            }
        }
    }
}

// Pack W[N][K] (f32, row-major, so GEMM computes x@W^T) into f16 WMMA-B tiles:
// ((nt*(K/32) + kt)*32 + lane)*16 + e, zero-padded to Npad rows.
__global__ void pack_b_kernel(const float* __restrict__ W, _Float16* __restrict__ Bp,
                              int N, int K, int Npad)
{
    const int total = Npad * K;
    int i = blockIdx.x * TPB + threadIdx.x;
    if (i >= total) return;
    const int e = i & 15;
    int t = i >> 4;
    const int lane = t & 31; t >>= 5;
    const int kTiles = K >> 5;
    const int kt = t % kTiles;
    const int nt = t / kTiles;
    const int n = nt * 16 + (lane & 15);
    const int k = kt * 32 + (lane >> 4) * 16 + e;
    _Float16 v = (_Float16)0.0f;
    if (n < N) v = (_Float16)W[(size_t)n * K + k];
    Bp[i] = v;
}

// ---------------------------------------------------------------------------
// Direct 3x3 conv, pad 1, stride (sh,sw), NCHW, f32 (tiny share of total work)
// ---------------------------------------------------------------------------
__global__ void conv3x3_kernel(const float* __restrict__ in, const float* __restrict__ w,
                               float* __restrict__ out, int Cin, int Cout,
                               int Hin, int Win, int Hout, int Wout, int sh, int sw,
                               int total)
{
    int idx = blockIdx.x * TPB + threadIdx.x;
    if (idx >= total) return;
    const int x  = idx % Wout;
    const int y  = (idx / Wout) % Hout;
    const int co = (idx / (Wout * Hout)) % Cout;
    const int b  = idx / (Wout * Hout * Cout);
    float s = 0.0f;
    for (int ci = 0; ci < Cin; ++ci) {
        const float* ip = in + ((size_t)(b * Cin + ci) * Hin) * Win;
        const float* wp = w + ((size_t)(co * Cin + ci) * 9);
        #pragma unroll
        for (int ky = 0; ky < 3; ++ky) {
            const int iy = y * sh + ky - 1;
            if (iy < 0 || iy >= Hin) continue;
            #pragma unroll
            for (int kx = 0; kx < 3; ++kx) {
                const int ix = x * sw + kx - 1;
                if (ix < 0 || ix >= Win) continue;
                s += ip[(size_t)iy * Win + ix] * wp[ky * 3 + kx];
            }
        }
    }
    out[idx] = s;
}

// Per-channel batch stats (mean, var) over (B, H, W); one block per channel.
__global__ void bnstat_kernel(const float* __restrict__ x, float* __restrict__ stat,
                              int C, int HW)
{
    const int c = blockIdx.x;
    const int tid = threadIdx.x;
    float s = 0.0f, s2 = 0.0f;
    const int n = 32 * HW;
    for (int t = tid; t < n; t += TPB) {
        const int b = t / HW, i = t % HW;
        const float v = x[((size_t)b * C + c) * HW + i];
        s += v; s2 += v * v;
    }
    __shared__ float rs[TPB], rq[TPB];
    rs[tid] = s; rq[tid] = s2;
    __syncthreads();
    for (int st = TPB / 2; st > 0; st >>= 1) {
        if (tid < st) { rs[tid] += rs[tid + st]; rq[tid] += rq[tid + st]; }
        __syncthreads();
    }
    if (tid == 0) {
        const float mean = rs[0] / (float)n;
        stat[c] = mean;
        stat[C + c] = rq[0] / (float)n - mean * mean;
    }
}

__global__ void bn_elu_kernel(float* __restrict__ x, const float* __restrict__ stat,
                              const float* __restrict__ g, const float* __restrict__ bb,
                              int C, int HW, int total, int doBN)
{
    int idx = blockIdx.x * TPB + threadIdx.x;
    if (idx >= total) return;
    float v = x[idx];
    if (doBN) {
        const int c = (idx / HW) % C;
        v = (v - stat[c]) * rsqrtf(stat[C + c] + 1e-5f) * g[c] + bb[c];
    }
    x[idx] = (v > 0.0f) ? v : (__expf(v) - 1.0f);   // ELU
}

// xs0[t][b][c*10+w] = conv4_out[b][c][t][w]  (f32 -> f16)
__global__ void build_xs0_kernel(const float* __restrict__ c4, _Float16* __restrict__ xs,
                                 int total)
{
    int idx = blockIdx.x * TPB + threadIdx.x;
    if (idx >= total) return;
    const int f = idx % 1280;
    const int b = (idx / 1280) % 32;
    const int t = idx / (1280 * 32);
    const int c = f / 10, w = f % 10;
    xs[idx] = (_Float16)c4[(((size_t)b * 128 + c) * 512 + t) * 10 + w];
}

// GRU gate fusion: h = (1-z)*tanh(in + r*hn) + z*h  ; writes f32 h, f16 h,
// and optionally the f16 sequence slot consumed by the next layer's big GEMM.
__global__ void gru_pointwise_kernel(const float* __restrict__ gx, const float* __restrict__ gh,
                                     const float* __restrict__ hprev, float* __restrict__ hnew,
                                     _Float16* __restrict__ hnew_h, _Float16* __restrict__ seq,
                                     int H, int seqStride)
{
    int idx = blockIdx.x * TPB + threadIdx.x;
    if (idx >= 32 * H) return;
    const int b = idx / H, j = idx % H;
    const float* gxb = gx + (size_t)b * 3 * H;
    const float* ghb = gh + (size_t)b * 3 * H;
    const float r = sigf_(gxb[j] + ghb[j]);
    const float z = sigf_(gxb[H + j] + ghb[H + j]);
    const float n = tanhf(gxb[2 * H + j] + r * ghb[2 * H + j]);
    const float h = (1.0f - z) * n + z * hprev[idx];
    hnew[idx] = h;
    hnew_h[idx] = (_Float16)h;
    if (seq) seq[(size_t)b * seqStride + j] = (_Float16)h;
}

__global__ void f16_sigmoid_kernel(const _Float16* __restrict__ in,
                                   _Float16* __restrict__ out, int n)
{
    int i = blockIdx.x * TPB + threadIdx.x;
    if (i >= n) return;
    out[i] = (_Float16)sigf_((float)in[i]);
}

__global__ void bcast_row_kernel(const float* __restrict__ src, float* __restrict__ dstf,
                                 _Float16* __restrict__ dsth, int H)
{
    int idx = blockIdx.x * TPB + threadIdx.x;
    if (idx >= 32 * H) return;
    const float v = src[idx % H];
    dstf[idx] = v;
    dsth[idx] = (_Float16)v;
}

// Attention (computed once): energy[b,s]=q[b]·keys[s,b]; softmax over s; ctx.
// kv layout: [(s*32+b)][384] = keys(128) | values(256). One block per b.
__global__ void attention_kernel(const float* __restrict__ q, const float* __restrict__ kv,
                                 float* __restrict__ ctx)
{
    const int b = blockIdx.x, tid = threadIdx.x;
    __shared__ float e[512];
    __shared__ float red[TPB];
    const float* qp = q + b * 128;
    for (int s = tid; s < 512; s += TPB) {
        const float* kp = kv + ((size_t)s * 32 + b) * 384;
        float d = 0.0f;
        for (int k = 0; k < 128; ++k) d += qp[k] * kp[k];
        e[s] = d;
    }
    __syncthreads();
    float m = -1e30f;
    for (int s = tid; s < 512; s += TPB) m = fmaxf(m, e[s]);
    red[tid] = m; __syncthreads();
    for (int st = TPB / 2; st > 0; st >>= 1) {
        if (tid < st) red[tid] = fmaxf(red[tid], red[tid + st]);
        __syncthreads();
    }
    m = red[0]; __syncthreads();
    float ss = 0.0f;
    for (int s = tid; s < 512; s += TPB) { const float ex = __expf(e[s] - m); e[s] = ex; ss += ex; }
    red[tid] = ss; __syncthreads();
    for (int st = TPB / 2; st > 0; st >>= 1) {
        if (tid < st) red[tid] += red[tid + st];
        __syncthreads();
    }
    const float inv = 1.0f / red[0];
    __syncthreads();
    float acc = 0.0f;   // tid == value index v (256 values)
    for (int s = 0; s < 512; ++s) acc += e[s] * kv[((size_t)s * 32 + b) * 384 + 128 + tid];
    ctx[b * 256 + tid] = acc * inv;
}

// first_h = concat(ctx[256], first_hidden[512]) -> 32 x 768 (f32 + f16)
__global__ void build_firsth_kernel(const float* __restrict__ ctx, const float* __restrict__ fh,
                                    float* __restrict__ hf, _Float16* __restrict__ hh)
{
    int idx = blockIdx.x * TPB + threadIdx.x;
    if (idx >= 32 * 768) return;
    const int b = idx / 768, j = idx % 768;
    const float v = (j < 256) ? ctx[b * 256 + j] : fh[j - 256];
    hf[idx] = v;
    hh[idx] = (_Float16)v;
}

// Decoder cell0 input GEMM = one-hot row gather: gx[b][n] = Wih[n][tgt[b][t]] + bih[n]
__global__ void dec_gx0_kernel(const float* __restrict__ Wih, const float* __restrict__ bih,
                               const int* __restrict__ target, int t, float* __restrict__ gx)
{
    int idx = blockIdx.x * TPB + threadIdx.x;
    if (idx >= 32 * 2304) return;
    const int b = idx / 2304, n = idx % 2304;
    const int ch = target[b * 256 + t];
    gx[idx] = Wih[(size_t)n * 33 + ch] + bih[n];
}

// tmp[(t*32+b)][48] -> out[b][t][c]  (B=32, T=256, C=33)
__global__ void out_reorder_kernel(const float* __restrict__ tmp, float* __restrict__ out)
{
    int idx = blockIdx.x * TPB + threadIdx.x;
    if (idx >= 32 * 256 * 33) return;
    const int c = idx % 33;
    const int t = (idx / 33) % 256;
    const int b = idx / (33 * 256);
    out[idx] = tmp[((size_t)t * 32 + b) * 48 + c];
}

// ---------------------------------------------------------------------------
// Host orchestration
// ---------------------------------------------------------------------------
static inline int gblk(long long n) { return (int)((n + TPB - 1) / TPB); }

extern "C" void kernel_launch(void* const* d_in, const int* in_sizes, int n_in,
                              void* d_out, int out_size, void* d_ws, size_t ws_size,
                              hipStream_t stream)
{
    (void)in_sizes; (void)n_in; (void)out_size; (void)ws_size;
    // Flattened input order (insertion order, depth-first through the dicts):
    //  0:x 1:target 2..5:convs 6..11:bn{g,b}x3
    //  12..35: gru[l][dir]{Wih,Whh,bih,bhh}  (l0d0=12, l0d1=16, l1d0=20, l1d1=24, l2d0=28, l2d1=32)
    //  36:enc_h0 37:enc_score_W 38:enc_score_b
    //  39..54: dec cells {Wih,Whh,bih,bhh}x4  55..58: dec h0[0..3]
    //  59:first_hidden 60:qW 61:qb 62:dec_score_W 63:dec_score_b
    auto F = [&](int i) -> const float* { return (const float*)d_in[i]; };
    const int* target = (const int*)d_in[1];
    float* out = (float*)d_out;

    // ---- workspace bump allocator ----
    char* base = (char*)d_ws;
    size_t used = 0;
    auto alloc = [&](size_t bytes) -> void* {
        size_t off = (used + 255) & ~(size_t)255;
        used = off + bytes;
        return (void*)(base + off);
    };
    // conv scratch region, unioned with the hoisted-GEMM buffer GX (dead by then)
    const size_t c1B = (size_t)32 * 16 * 1024 * 20 * 4;   // 41.9 MB
    const size_t c2B = (size_t)32 * 32 * 1024 * 20 * 4;   // 83.9 MB
    char* region = (char*)alloc(c1B + c2B);
    float* c1 = (float*)region;            // conv1 out, later conv3 out
    float* c2 = (float*)(region + c1B);    // conv2 out, later conv4 out
    float* c3 = c1;
    float* c4 = c2;
    float* GX = (float*)region;            // 16384 x 1536 f32 = 100.7 MB <= region

    float*    bnstat = (float*)alloc(2 * 128 * 4);
    _Float16* xs0h   = (_Float16*)alloc((size_t)512 * 32 * 1280 * 2);
    _Float16* xsA    = (_Float16*)alloc((size_t)512 * 32 * 1024 * 2);
    _Float16* xsB    = (_Float16*)alloc((size_t)512 * 32 * 1024 * 2);
    _Float16* sigh   = (_Float16*)alloc((size_t)512 * 32 * 1024 * 2);
    float*    kv     = (float*)alloc((size_t)16384 * 384 * 4);
    _Float16* pwih   = (_Float16*)alloc((size_t)1536 * 1280 * 2);   // enc pack scratch
    _Float16* pwhh   = (_Float16*)alloc((size_t)1536 * 512 * 2);
    float*    GH     = (float*)alloc((size_t)32 * 2304 * 4);
    float*    GXd    = (float*)alloc((size_t)32 * 2304 * 4);
    float*    henc   = (float*)alloc((size_t)32 * 512 * 4);
    _Float16* hench  = (_Float16*)alloc((size_t)32 * 512 * 2);
    float*    q      = (float*)alloc((size_t)32 * 128 * 4);
    float*    qinf   = (float*)alloc((size_t)32 * 512 * 4);
    _Float16* qinh   = (_Float16*)alloc((size_t)32 * 512 * 2);
    float*    ctx    = (float*)alloc((size_t)32 * 256 * 4);
    float*    hA     = (float*)alloc((size_t)32 * 768 * 4);
    _Float16* hAh    = (_Float16*)alloc((size_t)32 * 768 * 2);
    float*    h1     = (float*)alloc((size_t)32 * 512 * 4);
    _Float16* h1h    = (_Float16*)alloc((size_t)32 * 512 * 2);
    float*    h2     = (float*)alloc((size_t)32 * 512 * 4);
    _Float16* h2h    = (_Float16*)alloc((size_t)32 * 512 * 2);
    float*    h3     = (float*)alloc((size_t)32 * 512 * 4);
    _Float16* h3h    = (_Float16*)alloc((size_t)32 * 512 * 2);
    _Float16* h3seq  = (_Float16*)alloc((size_t)8192 * 512 * 2);
    _Float16* h3sig  = (_Float16*)alloc((size_t)8192 * 512 * 2);
    _Float16* pesc   = (_Float16*)alloc((size_t)384 * 1024 * 2);
    _Float16* pqW    = (_Float16*)alloc((size_t)128 * 512 * 2);
    _Float16* pWhh0  = (_Float16*)alloc((size_t)2304 * 768 * 2);
    _Float16* pWih1  = (_Float16*)alloc((size_t)1536 * 768 * 2);
    _Float16* pWhh1  = (_Float16*)alloc((size_t)1536 * 512 * 2);
    _Float16* pWih2  = (_Float16*)alloc((size_t)1536 * 512 * 2);
    _Float16* pWhh2  = (_Float16*)alloc((size_t)1536 * 512 * 2);
    _Float16* pWih3  = (_Float16*)alloc((size_t)1536 * 512 * 2);
    _Float16* pWhh3  = (_Float16*)alloc((size_t)1536 * 512 * 2);
    _Float16* pdsc   = (_Float16*)alloc((size_t)48 * 512 * 2);
    float*    sctmp  = (float*)alloc((size_t)8192 * 48 * 4);

    auto gemm = [&](const _Float16* A, const _Float16* Bp, const float* bias, float* C,
                    int M, int N, int Npad, int K, int ldc) {
        dim3 grid((Npad + 127) / 128, M / 32);
        wmma_gemm_kernel<<<grid, TPB, 0, stream>>>(A, Bp, bias, C, M, N, Npad, K, ldc);
    };
    auto pack = [&](const float* W, _Float16* Bp, int N, int K, int Npad) {
        pack_b_kernel<<<gblk((long long)Npad * K), TPB, 0, stream>>>(W, Bp, N, K, Npad);
    };

    // ================= Encoder conv stack =================
    {
        int tot;
        tot = 32 * 16 * 1024 * 20;
        conv3x3_kernel<<<gblk(tot), TPB, 0, stream>>>(F(0), F(2), c1, 1, 16, 1024, 40, 1024, 20, 1, 2, tot);
        bnstat_kernel<<<16, TPB, 0, stream>>>(c1, bnstat, 16, 1024 * 20);
        bn_elu_kernel<<<gblk(tot), TPB, 0, stream>>>(c1, bnstat, F(6), F(7), 16, 1024 * 20, tot, 1);

        tot = 32 * 32 * 1024 * 20;
        conv3x3_kernel<<<gblk(tot), TPB, 0, stream>>>(c1, F(3), c2, 16, 32, 1024, 20, 1024, 20, 1, 1, tot);
        bnstat_kernel<<<32, TPB, 0, stream>>>(c2, bnstat, 32, 1024 * 20);
        bn_elu_kernel<<<gblk(tot), TPB, 0, stream>>>(c2, bnstat, F(8), F(9), 32, 1024 * 20, tot, 1);

        tot = 32 * 64 * 512 * 10;
        conv3x3_kernel<<<gblk(tot), TPB, 0, stream>>>(c2, F(4), c3, 32, 64, 1024, 20, 512, 10, 2, 2, tot);
        bn_elu_kernel<<<gblk(tot), TPB, 0, stream>>>(c3, bnstat, F(6), F(7), 64, 512 * 10, tot, 0); // ELU only

        tot = 32 * 128 * 512 * 10;
        conv3x3_kernel<<<gblk(tot), TPB, 0, stream>>>(c3, F(5), c4, 64, 128, 512, 10, 512, 10, 1, 1, tot);
        bnstat_kernel<<<128, TPB, 0, stream>>>(c4, bnstat, 128, 512 * 10);
        bn_elu_kernel<<<gblk(tot), TPB, 0, stream>>>(c4, bnstat, F(10), F(11), 128, 512 * 10, tot, 1);

        build_xs0_kernel<<<gblk(512LL * 32 * 1280), TPB, 0, stream>>>(c4, xs0h, 512 * 32 * 1280);
    }

    // ================= Encoder BiGRU stack =================
    const int Kin[3] = {1280, 1024, 1024};
    const _Float16* xs_in = xs0h;
    _Float16* xs_out = xsA;
    for (int l = 0; l < 3; ++l) {
        for (int dir = 0; dir < 2; ++dir) {
            const int wb = 12 + l * 8 + dir * 4;
            pack(F(wb + 0), pwih, 1536, Kin[l], 1536);
            pack(F(wb + 1), pwhh, 1536, 512, 1536);
            // hoisted input GEMM for all 512 steps: [16384 x Kin] @ Wih^T
            gemm(xs_in, pwih, F(wb + 2), GX, 512 * 32, 1536, 1536, Kin[l], 1536);
            bcast_row_kernel<<<gblk(32 * 512), TPB, 0, stream>>>(F(36) + (2 * l + dir) * 512,
                                                                 henc, hench, 512);
            for (int s = 0; s < 512; ++s) {
                const int t = dir ? (511 - s) : s;
                gemm(hench, pwhh, F(wb + 3), GH, 32, 1536, 1536, 512, 1536);
                gru_pointwise_kernel<<<gblk(32 * 512), TPB, 0, stream>>>(
                    GX + (size_t)t * 32 * 1536, GH, henc, henc, hench,
                    xs_out + (size_t)t * 32 * 1024 + dir * 512, 512, 1024);
            }
        }
        xs_in = xs_out;
        xs_out = (xs_out == xsA) ? xsB : xsA;
    }
    const _Float16* xs_fin = xs_in;   // xsA after 3 layers

    // ================= Encoder score -> keys|values =================
    f16_sigmoid_kernel<<<gblk(16384LL * 1024), TPB, 0, stream>>>(xs_fin, sigh, 16384 * 1024);
    pack(F(37), pesc, 384, 1024, 384);
    gemm(sigh, pesc, F(38), kv, 16384, 384, 384, 1024, 384);

    // ================= Attention (computed once) =================
    bcast_row_kernel<<<gblk(32 * 512), TPB, 0, stream>>>(F(58), qinf, qinh, 512);
    pack(F(60), pqW, 128, 512, 128);
    gemm(qinh, pqW, F(61), q, 32, 128, 128, 512, 128);
    attention_kernel<<<32, TPB, 0, stream>>>(q, kv, ctx);
    build_firsth_kernel<<<gblk(32 * 768), TPB, 0, stream>>>(ctx, F(59), hA, hAh);
    bcast_row_kernel<<<gblk(32 * 512), TPB, 0, stream>>>(F(56), h1, h1h, 512);
    bcast_row_kernel<<<gblk(32 * 512), TPB, 0, stream>>>(F(57), h2, h2h, 512);
    bcast_row_kernel<<<gblk(32 * 512), TPB, 0, stream>>>(F(58), h3, h3h, 512);

    // ================= Decoder weight packing =================
    pack(F(40), pWhh0, 2304, 768, 2304);
    pack(F(43), pWih1, 1536, 768, 1536);
    pack(F(44), pWhh1, 1536, 512, 1536);
    pack(F(47), pWih2, 1536, 512, 1536);
    pack(F(48), pWhh2, 1536, 512, 1536);
    pack(F(51), pWih3, 1536, 512, 1536);
    pack(F(52), pWhh3, 1536, 512, 1536);
    pack(F(62), pdsc, 33, 512, 48);

    // ================= Decoder 4-cell GRU scan =================
    for (int t = 0; t < 256; ++t) {
        // cell 0: one-hot input -> row gather; hidden 768
        dec_gx0_kernel<<<gblk(32 * 2304), TPB, 0, stream>>>(F(39), F(41), target, t, GXd);
        gemm(hAh, pWhh0, F(42), GH, 32, 2304, 2304, 768, 2304);
        gru_pointwise_kernel<<<gblk(32 * 768), TPB, 0, stream>>>(GXd, GH, hA, hA, hAh,
                                                                 (_Float16*)nullptr, 768, 0);
        // cell 1: input = cell0 h (768)
        gemm(hAh, pWih1, F(45), GXd, 32, 1536, 1536, 768, 1536);
        gemm(h1h, pWhh1, F(46), GH, 32, 1536, 1536, 512, 1536);
        gru_pointwise_kernel<<<gblk(32 * 512), TPB, 0, stream>>>(GXd, GH, h1, h1, h1h,
                                                                 (_Float16*)nullptr, 512, 0);
        // cell 2
        gemm(h1h, pWih2, F(49), GXd, 32, 1536, 1536, 512, 1536);
        gemm(h2h, pWhh2, F(50), GH, 32, 1536, 1536, 512, 1536);
        gru_pointwise_kernel<<<gblk(32 * 512), TPB, 0, stream>>>(GXd, GH, h2, h2, h2h,
                                                                 (_Float16*)nullptr, 512, 0);
        // cell 3 (writes h3 sequence)
        gemm(h2h, pWih3, F(53), GXd, 32, 1536, 1536, 512, 1536);
        gemm(h3h, pWhh3, F(54), GH, 32, 1536, 1536, 512, 1536);
        gru_pointwise_kernel<<<gblk(32 * 512), TPB, 0, stream>>>(GXd, GH, h3, h3, h3h,
                                                                 h3seq + (size_t)t * 32 * 512, 512, 512);
    }

    // ================= Decoder output projection =================
    f16_sigmoid_kernel<<<gblk(8192LL * 512), TPB, 0, stream>>>(h3seq, h3sig, 8192 * 512);
    gemm(h3sig, pdsc, F(63), sctmp, 8192, 33, 48, 512, 48);
    out_reorder_kernel<<<gblk(32 * 256 * 33), TPB, 0, stream>>>(sctmp, out);
}
